// CMPGNN_91207925498529
// MI455X (gfx1250) — matrix operand order repaired and checked by reading, not compiled, not run on hardware
//
#include <hip/hip_runtime.h>
#include <hip/hip_bf16.h>
#include <math.h>

// ---------------------------------------------------------------------------
// CMPGNN on MI455X (gfx1250): bf16 WMMA GEMMs fed by raw b128 loads
// (pre-converted operands), fused 3-way layer GEMM, wave-per-edge scatter.
// ---------------------------------------------------------------------------

typedef __attribute__((ext_vector_type(16))) __bf16 v16bf;
typedef __attribute__((ext_vector_type(8)))  __bf16 v8bf;
typedef __attribute__((ext_vector_type(8)))  float  v8f;

#define H_DIM 64

__device__ __forceinline__ __bf16 f2bf(float f) { return (__bf16)f; }

// ---- bf16 fragment loaders (no conversion; pure b128 loads) ---------------
// A: 16x32 tile, row-major bf16 [*, ld]. Lane m = lane%16;
// elems 0..7 -> K = k0+sel.., elems 8..15 -> K = k0+16+sel.., sel = lane<16?0:8.
__device__ __forceinline__ v16bf load_fragA_bf16(const __bf16* __restrict__ base,
                                                 int m0, int k0, int ld) {
    const int lane = threadIdx.x & 31;
    const int r    = m0 + (lane & 15);
    const int sel  = (lane & 16) ? 8 : 0;
    const __bf16* p = base + (size_t)r * ld + k0 + sel;
    v8bf lo = *(const v8bf*)(p);
    v8bf hi = *(const v8bf*)(p + 16);
    v16bf out;
#pragma unroll
    for (int i = 0; i < 8; ++i) { out[i] = lo[i]; out[i + 8] = hi[i]; }
    return out;
}

// B for Out = X @ W^T: B[k][n] = W[n][k]; lane n = lane%16 holds
// elems 0..7 -> K = k0+sel.., elems 8..15 -> K = k0+16+sel.., sel = lane<16?0:8. W row-major bf16 [64, ld].
__device__ __forceinline__ v16bf load_fragB_bf16(const __bf16* __restrict__ w,
                                                 int n0, int k0, int ld) {
    const int lane = threadIdx.x & 31;
    const int n    = n0 + (lane & 15);
    const int kb   = k0 + ((lane & 16) ? 8 : 0);
    const __bf16* p = w + (size_t)n * ld + kb;
    v8bf lo = *(const v8bf*)(p);
    v8bf hi = *(const v8bf*)(p + 16);
    v16bf out;
#pragma unroll
    for (int i = 0; i < 8; ++i) { out[i] = lo[i]; out[i + 8] = hi[i]; }
    return out;
}

// A from fp32 source (used only for x, which is read exactly once).
__device__ __forceinline__ v16bf load_fragA_f32(const float* __restrict__ base,
                                                int m0, int k0, int ld) {
    const int lane = threadIdx.x & 31;
    const int r    = m0 + (lane & 15);
    const int sel  = (lane & 16) ? 8 : 0;
    const float* p0 = base + (size_t)r * ld + k0 + sel;
    const float* p1 = p0 + 16;
    __builtin_prefetch(p0 + 32, 0, 1);     // next K-chunk -> global_prefetch_b8
    v16bf out;
#pragma unroll
    for (int i = 0; i < 8; ++i) {
        out[i]     = f2bf(p0[i]);
        out[i + 8] = f2bf(p1[i]);
    }
    return out;
}

__global__ __launch_bounds__(256)
void convert_f32_bf16(const float* __restrict__ src, __bf16* __restrict__ dst, int n) {
    int i = blockIdx.x * blockDim.x + threadIdx.x;
    if (i < n) dst[i] = f2bf(src[i]);
}

// ---- input projection: Q = x @ W_in^T + b_in ------------------------------
// One wave per 16-row tile, 4 col tiles; writes f32 Q and bf16 mirror Qb.
__global__ __launch_bounds__(256)
void wmma_gemm_in(const float* __restrict__ X, const __bf16* __restrict__ Wb,
                  const float* __restrict__ bias, float* __restrict__ Qf,
                  __bf16* __restrict__ Qb, int nrows, int K) {
    const int lane = threadIdx.x & 31;
    const int wave = threadIdx.x >> 5;
    const int tile = blockIdx.x * (blockDim.x >> 5) + wave;
    const int m0   = tile * 16;
    if (m0 >= nrows) return;

    v8f acc[4];
#pragma unroll
    for (int t = 0; t < 4; ++t)
#pragma unroll
        for (int v = 0; v < 8; ++v) acc[t][v] = 0.0f;

    for (int kb = 0; kb < K; kb += 32) {
        v16bf a = load_fragA_f32(X, m0, kb, K);
#pragma unroll
        for (int t = 0; t < 4; ++t) {
            v16bf b = load_fragB_bf16(Wb, t * 16, kb, K);
            acc[t] = __builtin_amdgcn_wmma_f32_16x16x32_bf16(
                false, a, false, b, (short)0, acc[t], false, false);
        }
    }

    const int n  = lane & 15;
    const int mb = m0 + ((lane & 16) ? 8 : 0);
#pragma unroll
    for (int t = 0; t < 4; ++t) {
        const int nn = t * 16 + n;
        const float bv = bias[nn];
#pragma unroll
        for (int v = 0; v < 8; ++v) {
            const float x = acc[t][v] + bv;
            const size_t idx = (size_t)(mb + v) * H_DIM + nn;
            Qf[idx] = x;
            Qb[idx] = f2bf(x);
        }
    }
}

// ---- fused per-layer GEMMs: h3 = Q W1^T, h4 = Q W2^T, hl = relu(Q Wf^T) ---
// A fragment loaded once per K-chunk, reused by 12 WMMAs.
__global__ __launch_bounds__(256)
void wmma_layer3(const __bf16* __restrict__ Qb,
                 const __bf16* __restrict__ W1b, const __bf16* __restrict__ W2b,
                 const __bf16* __restrict__ Wfb,
                 float* __restrict__ h3, float* __restrict__ h4,
                 float* __restrict__ hl, int nrows) {
    const int lane = threadIdx.x & 31;
    const int wave = threadIdx.x >> 5;
    const int tile = blockIdx.x * (blockDim.x >> 5) + wave;
    const int m0   = tile * 16;
    if (m0 >= nrows) return;

    v8f acc[3][4];
#pragma unroll
    for (int o = 0; o < 3; ++o)
#pragma unroll
        for (int t = 0; t < 4; ++t)
#pragma unroll
            for (int v = 0; v < 8; ++v) acc[o][t][v] = 0.0f;

#pragma unroll
    for (int kb = 0; kb < H_DIM; kb += 32) {
        v16bf a = load_fragA_bf16(Qb, m0, kb, H_DIM);
#pragma unroll
        for (int t = 0; t < 4; ++t) {
            v16bf b1 = load_fragB_bf16(W1b, t * 16, kb, H_DIM);
            acc[0][t] = __builtin_amdgcn_wmma_f32_16x16x32_bf16(
                false, a, false, b1, (short)0, acc[0][t], false, false);
            v16bf b2 = load_fragB_bf16(W2b, t * 16, kb, H_DIM);
            acc[1][t] = __builtin_amdgcn_wmma_f32_16x16x32_bf16(
                false, a, false, b2, (short)0, acc[1][t], false, false);
            v16bf bf_ = load_fragB_bf16(Wfb, t * 16, kb, H_DIM);
            acc[2][t] = __builtin_amdgcn_wmma_f32_16x16x32_bf16(
                false, a, false, bf_, (short)0, acc[2][t], false, false);
        }
    }

    const int n  = lane & 15;
    const int mb = m0 + ((lane & 16) ? 8 : 0);
#pragma unroll
    for (int t = 0; t < 4; ++t) {
        const int nn = t * 16 + n;
#pragma unroll
        for (int v = 0; v < 8; ++v) {
            const size_t idx = (size_t)(mb + v) * H_DIM + nn;
            h3[idx] = acc[0][t][v];
            h4[idx] = acc[1][t][v];
            hl[idx] = fmaxf(acc[2][t][v], 0.0f);
        }
    }
}

__global__ __launch_bounds__(256)
void fill_zero(float* __restrict__ p, size_t n) {
    size_t i = (size_t)blockIdx.x * blockDim.x + threadIdx.x;
    size_t stride = (size_t)gridDim.x * blockDim.x;
    for (; i < n; i += stride) p[i] = 0.0f;
}

// ---- wave-per-edge gather / gated scatter-add -----------------------------
__global__ __launch_bounds__(256)
void edge_kernel(const float* __restrict__ h3, const float* __restrict__ h4,
                 const int* __restrict__ row, const int* __restrict__ col,
                 float* __restrict__ agg3, float* __restrict__ agg4, int nedges) {
    const int wid  = (int)(((size_t)blockIdx.x * blockDim.x + threadIdx.x) >> 5);
    const int lane = threadIdx.x & 31;
    if (wid >= nedges) return;

    const int r = row[wid];
    const int c = col[wid];
    const float* h3r = h3 + (size_t)r * H_DIM;
    const float* h4c = h4 + (size_t)c * H_DIM;
    const float* h4r = h4 + (size_t)r * H_DIM;

    const float a0 = h3r[lane], a1 = h3r[lane + 32];
    const float b0 = h4c[lane], b1 = h4c[lane + 32];
    float d = a0 * b0 + a1 * b1;
#pragma unroll
    for (int off = 16; off > 0; off >>= 1) d += __shfl_xor(d, off, 32);

    const float s  = 1.0f / (1.0f + __expf(d));   // sigmoid(-d)
    const float t  = 1.0f - s;
    const float c0 = h4r[lane], c1 = h4r[lane + 32];

    float* A3 = agg3 + (size_t)c * H_DIM;
    float* A4 = agg4 + (size_t)c * H_DIM;
    atomicAdd(&A3[lane],      s * a0);
    atomicAdd(&A3[lane + 32], s * a1);
    atomicAdd(&A4[lane],      t * c0);
    atomicAdd(&A4[lane + 32], t * c1);
}

// ---- Q = L2-normalize(hl + agg3 - agg4); writes f32 Q and bf16 mirror -----
__global__ __launch_bounds__(256)
void combine_norm(const float* __restrict__ hloop, const float* __restrict__ agg3,
                  const float* __restrict__ agg4, float* __restrict__ Q,
                  __bf16* __restrict__ Qb, int n) {
    const int wid  = (int)(((size_t)blockIdx.x * blockDim.x + threadIdx.x) >> 5);
    const int lane = threadIdx.x & 31;
    if (wid >= n) return;
    const size_t base = (size_t)wid * H_DIM;

    const float x0 = hloop[base + lane]      + agg3[base + lane]      - agg4[base + lane];
    const float x1 = hloop[base + lane + 32] + agg3[base + lane + 32] - agg4[base + lane + 32];
    float ss = x0 * x0 + x1 * x1;
#pragma unroll
    for (int off = 16; off > 0; off >>= 1) ss += __shfl_xor(ss, off, 32);
    const float inv = 1.0f / fmaxf(sqrtf(ss), 1e-12f);
    const float y0 = x0 * inv, y1 = x1 * inv;
    Q[base + lane]       = y0;
    Q[base + lane + 32]  = y1;
    Qb[base + lane]      = f2bf(y0);
    Qb[base + lane + 32] = f2bf(y1);
}

// ---- logits + log_softmax (C = 40), one wave per node ---------------------
__global__ __launch_bounds__(256)
void out_kernel(const float* __restrict__ Q, const float* __restrict__ Wout,
                const float* __restrict__ bout, float* __restrict__ out,
                int n, int nclass) {
    const int wid  = (int)(((size_t)blockIdx.x * blockDim.x + threadIdx.x) >> 5);
    const int lane = threadIdx.x & 31;
    if (wid >= n) return;
    const size_t qb = (size_t)wid * H_DIM;

    const int  c0 = lane;
    const int  c1 = lane + 32;
    const bool v1 = (c1 < nclass);

    float acc0 = bout[c0];
    float acc1 = v1 ? bout[c1] : 0.0f;
#pragma unroll 8
    for (int k = 0; k < H_DIM; ++k) {
        const float qv = Q[qb + k];                 // uniform addr: broadcast load
        acc0 += qv * Wout[(size_t)c0 * H_DIM + k];
        if (v1) acc1 += qv * Wout[(size_t)c1 * H_DIM + k];
    }

    float m = fmaxf(acc0, v1 ? acc1 : -INFINITY);
#pragma unroll
    for (int off = 16; off > 0; off >>= 1) m = fmaxf(m, __shfl_xor(m, off, 32));
    float e = __expf(acc0 - m) + (v1 ? __expf(acc1 - m) : 0.0f);
#pragma unroll
    for (int off = 16; off > 0; off >>= 1) e += __shfl_xor(e, off, 32);
    const float lse = m + __logf(e);

    out[(size_t)wid * nclass + c0] = acc0 - lse;
    if (v1) out[(size_t)wid * nclass + c1] = acc1 - lse;
}

// ---------------------------------------------------------------------------
extern "C" void kernel_launch(void* const* d_in, const int* in_sizes, int n_in,
                              void* d_out, int out_size, void* d_ws, size_t ws_size,
                              hipStream_t stream) {
    const float* x    = (const float*)d_in[0];
    const int*   ei   = (const int*)  d_in[1];
    const float* Win  = (const float*)d_in[2];
    const float* bin  = (const float*)d_in[3];
    const float* Wout = (const float*)d_in[10];
    const float* bout = (const float*)d_in[11];

    const int F_IN = 512;
    const int N    = in_sizes[0] / F_IN;     // 100000
    const int E    = in_sizes[1] / 2;        // 1250000
    const int C    = out_size / N;           // 40

    const int* row = ei;
    const int* col = ei + E;

    // ---- workspace carve-out ------------------------------------------------
    const size_t per = (size_t)N * H_DIM;
    float* Q    = (float*)d_ws;          // 6 fp32 buffers of N*64
    float* h3   = Q    + per;
    float* h4   = h3   + per;
    float* hl   = h4   + per;
    float* agg3 = hl   + per;
    float* agg4 = agg3 + per;
    __bf16* Qb   = (__bf16*)(agg4 + per);                 // bf16 mirror of Q
    __bf16* Winb = Qb + per;                              // 64*512
    __bf16* Wb[6];                                        // 6x 64*64
    Wb[0] = Winb + (size_t)H_DIM * F_IN;
    for (int i = 1; i < 6; ++i) Wb[i] = Wb[i - 1] + H_DIM * H_DIM;

    const int WPB = 8;                                   // waves / 256-thread block
    const int gemm_blocks = ((N / 16) + WPB - 1) / WPB;
    const int edge_blocks = (E + WPB - 1) / WPB;
    const int node_blocks = (N + WPB - 1) / WPB;
    const int fill_blocks = (int)((per + 255) / 256);

    // ---- pre-convert all weights to bf16 (tiny) -----------------------------
    convert_f32_bf16<<<(H_DIM * F_IN + 255) / 256, 256, 0, stream>>>(Win, Winb, H_DIM * F_IN);
    for (int i = 0; i < 6; ++i)
        convert_f32_bf16<<<(H_DIM * H_DIM + 255) / 256, 256, 0, stream>>>(
            (const float*)d_in[4 + i], Wb[i], H_DIM * H_DIM);

    // ---- Q = x @ W_in^T + b_in ---------------------------------------------
    wmma_gemm_in<<<gemm_blocks, 256, 0, stream>>>(x, Winb, bin, Q, Qb, N, F_IN);

    for (int layer = 0; layer < 2; ++layer) {
        wmma_layer3<<<gemm_blocks, 256, 0, stream>>>(
            Qb, Wb[layer * 3 + 0], Wb[layer * 3 + 1], Wb[layer * 3 + 2],
            h3, h4, hl, N);

        fill_zero<<<fill_blocks, 256, 0, stream>>>(agg3, per);
        fill_zero<<<fill_blocks, 256, 0, stream>>>(agg4, per);

        edge_kernel<<<edge_blocks, 256, 0, stream>>>(h3, h4, row, col, agg3, agg4, E);

        combine_norm<<<node_blocks, 256, 0, stream>>>(hl, agg3, agg4, Q, Qb, N);
    }

    out_kernel<<<node_blocks, 256, 0, stream>>>(Q, Wout, bout, (float*)d_out, N, C);
}
